// NAOCell_3169685865086
// MI455X (gfx1250) — compile-verified, matrix operand order checked by reading, not run on hardware
//
#include <hip/hip_runtime.h>
#include <hip/hip_bf16.h>

typedef __attribute__((ext_vector_type(16))) __bf16 v16bf;
typedef __attribute__((ext_vector_type(8)))  __bf16 v8bf;
typedef __attribute__((ext_vector_type(8)))  float  v8f;

#define T_STEPS 128
#define BATCH   256
#define NHID    850
#define NTOT    1700          // 2*NHID
#define KPAD    864           // 27*32
#define BM      128
#define BN      64
#define BK      32
#define LDT     40            // LDS row stride in bf16: 80B rows, 16B aligned

// ---------------------------------------------------------------------------
// Weights: f32 row-major [K][N] -> bf16 transposed padded [N][KPAD]
// slot 0: W0 x-part, slot 1: W0 h-part, slots 2..9: Ws[0..7]
// ---------------------------------------------------------------------------
__global__ void convert_weights_kernel(const float* __restrict__ W0,
                                       const float* __restrict__ Ws,
                                       __bf16* __restrict__ wT) {
    const long total = 10L * NTOT * KPAD;
    for (long i = (long)blockIdx.x * blockDim.x + threadIdx.x; i < total;
         i += (long)gridDim.x * blockDim.x) {
        int  k = (int)(i % KPAD);
        long r = i / KPAD;
        int  n = (int)(r % NTOT);
        int  w = (int)(r / NTOT);
        float v = 0.0f;
        if (k < NHID) {
            if (w == 0)      v = W0[(long)k * NTOT + n];
            else if (w == 1) v = W0[(long)(NHID + k) * NTOT + n];
            else             v = Ws[((long)(w - 2) * NHID + k) * NTOT + n];
        }
        wT[i] = (__bf16)v;
    }
}

// f32 [rows][NHID] -> bf16 [rows][KPAD] (zero pad); src==null -> all zeros
__global__ void cvt_pad_kernel(__bf16* __restrict__ dst,
                               const float* __restrict__ src, long rows) {
    const long total = rows * KPAD;
    for (long i = (long)blockIdx.x * blockDim.x + threadIdx.x; i < total;
         i += (long)gridDim.x * blockDim.x) {
        int  k = (int)(i % KPAD);
        long r = i / KPAD;
        float v = (src != nullptr && k < NHID) ? src[r * NHID + k] : 0.0f;
        dst[i] = (__bf16)v;
    }
}

// ---------------------------------------------------------------------------
// WMMA GEMM. MODE 0: plain C = A*B (input projection).
//           MODE 1: dual-half fused gate:
//             c = A*Bc (+add), h = A*Bh (+add); out = pred + sig(c)*(act(h)-pred)
// A: bf16 [M][KPAD] padded.  B: bf16 [Ncols][KPAD] transposed padded.
// ---------------------------------------------------------------------------
struct Prob {
    const __bf16* A;        // [M][KPAD]
    const __bf16* BTc;
    const __bf16* BTh;
    const float*  pred;     // [M][NHID]
    const float*  addend;   // [M][NTOT] or null
    float*        out;      // MODE1: [M][NHID]; MODE0: [M][Ncols]
    __bf16*       outb;     // MODE1: [M][KPAD] bf16 copy (pad already zeroed)
    int           act;      // 0 tanh, 1 relu, 2 sigmoid, 3 identity
};
struct ProbSet { Prob p[4]; };

__device__ __forceinline__ float sigmoidf_(float x) {
    return 1.0f / (1.0f + __expf(-x));
}

template <int MODE>
__launch_bounds__(256)
__global__ void gemm_wmma_kernel(ProbSet ps, int Ncols) {
    const Prob pb = ps.p[blockIdx.z];

    __shared__ __align__(16) __bf16 As[2][BM * LDT];
    __shared__ __align__(16) __bf16 Bs[2][2 * BN * LDT];

    const int tid  = threadIdx.x;
    const int lane = tid & 31;
    const int wave = tid >> 5;
    const int wm   = wave & 3;          // wave row block (32 rows)
    const int wn   = wave >> 2;         // wave col block (32 cols)
    const int m0   = blockIdx.x * BM;
    const int n0   = blockIdx.y * BN;
    const int half = lane >> 4;
    const int l16  = lane & 15;

    // staging addresses (all 16B aligned: KPAD*2=1728B row stride)
    const int ar = tid >> 1, ac = (tid & 1) * 16;       // A: 128 rows x 2 chunks
    const int bn = tid >> 2, bk = (tid & 3) * 8;        // B: 64 rows x 4 chunks
    int ng = n0 + bn; if (ng > Ncols - 1) ng = Ncols - 1;
    const __bf16* aptr  = pb.A   + (long)(m0 + ar) * KPAD + ac;
    const __bf16* bptrc = pb.BTc + (long)ng * KPAD + bk;
    const __bf16* bptrh = MODE ? (pb.BTh + (long)ng * KPAD + bk) : nullptr;

    v8bf ra0, ra1, rbc, rbh;
    auto load_tile = [&](int k0) {
        ra0 = *(const v8bf*)(aptr + k0);
        ra1 = *(const v8bf*)(aptr + k0 + 8);
        rbc = *(const v8bf*)(bptrc + k0);
        if (MODE) rbh = *(const v8bf*)(bptrh + k0);
    };
    auto store_tile = [&](int p) {
        *(v8bf*)&As[p][ar * LDT + ac]     = ra0;
        *(v8bf*)&As[p][ar * LDT + ac + 8] = ra1;
        *(v8bf*)&Bs[p][bn * LDT + bk]     = rbc;
        if (MODE) *(v8bf*)&Bs[p][BN * LDT + bn * LDT + bk] = rbh;
    };

    v8f cc[2][2], ch[2][2];
#pragma unroll
    for (int i = 0; i < 2; ++i)
#pragma unroll
        for (int j = 0; j < 2; ++j) { cc[i][j] = (v8f)0.0f; ch[i][j] = (v8f)0.0f; }

    load_tile(0);
    store_tile(0);
    __syncthreads();

    const int NT = KPAD / BK;   // 27
    for (int it = 0; it < NT; ++it) {
        const int p = it & 1;
        // issue next tile's global loads before compute (latency overlap)
        if (it + 1 < NT) load_tile((it + 1) * BK);
        if (it + 2 < NT) __builtin_prefetch(bptrc + (it + 2) * BK, 0, 1);

        // gather fragments from LDS (documented 16-bit A/B VGPR layouts)
        v16bf af[2], bcf[2], bhf[2];
#pragma unroll
        for (int mi = 0; mi < 2; ++mi) {
            const int row = wm * 32 + mi * 16 + l16;
            const v8bf lo = *(const v8bf*)&As[p][row * LDT + half * 8];
            const v8bf hi = *(const v8bf*)&As[p][row * LDT + 16 + half * 8];
#pragma unroll
            for (int e = 0; e < 8; ++e) { af[mi][e] = lo[e]; af[mi][8 + e] = hi[e]; }
        }
#pragma unroll
        for (int ni = 0; ni < 2; ++ni) {
            const int col = wn * 32 + ni * 16 + l16;
            const v8bf lo = *(const v8bf*)&Bs[p][col * LDT + half * 8];
            const v8bf hi = *(const v8bf*)&Bs[p][col * LDT + 16 + half * 8];
#pragma unroll
            for (int e = 0; e < 8; ++e) { bcf[ni][e] = lo[e]; bcf[ni][8 + e] = hi[e]; }
            if (MODE) {
                const v8bf lo2 = *(const v8bf*)&Bs[p][BN * LDT + col * LDT + half * 8];
                const v8bf hi2 = *(const v8bf*)&Bs[p][BN * LDT + col * LDT + 16 + half * 8];
#pragma unroll
                for (int e = 0; e < 8; ++e) { bhf[ni][e] = lo2[e]; bhf[ni][8 + e] = hi2[e]; }
            }
        }

#pragma unroll
        for (int mi = 0; mi < 2; ++mi)
#pragma unroll
            for (int ni = 0; ni < 2; ++ni) {
                cc[mi][ni] = __builtin_amdgcn_wmma_f32_16x16x32_bf16(
                    false, af[mi], false, bcf[ni], (short)0, cc[mi][ni], false, false);
                if (MODE)
                    ch[mi][ni] = __builtin_amdgcn_wmma_f32_16x16x32_bf16(
                        false, af[mi], false, bhf[ni], (short)0, ch[mi][ni], false, false);
            }

        // store next tile into the other LDS buffer; single barrier per iter
        if (it + 1 < NT) store_tile(p ^ 1);
        __syncthreads();
    }

    // ---- epilogue ----
#pragma unroll
    for (int mi = 0; mi < 2; ++mi)
#pragma unroll
        for (int ni = 0; ni < 2; ++ni)
#pragma unroll
            for (int r = 0; r < 8; ++r) {
                const int m = m0 + wm * 32 + mi * 16 + half * 8 + r;
                const int n = n0 + wn * 32 + ni * 16 + l16;
                float c = cc[mi][ni][r];
                if (MODE == 0) {
                    if (n < Ncols) pb.out[(long)m * Ncols + n] = c;
                } else {
                    float h = ch[mi][ni][r];
                    const int ncl = (n < NHID) ? n : NHID - 1;
                    if (pb.addend) {
                        c += pb.addend[(long)m * NTOT + ncl];
                        h += pb.addend[(long)m * NTOT + NHID + ncl];
                    }
                    const float pv = pb.pred[(long)m * NHID + ncl];
                    switch (pb.act) {
                        case 0: h = 2.0f * sigmoidf_(2.0f * h) - 1.0f; break; // tanh
                        case 1: h = fmaxf(h, 0.0f); break;
                        case 2: h = sigmoidf_(h); break;
                        default: break;
                    }
                    const float s = pv + sigmoidf_(c) * (h - pv);
                    if (n < NHID) {
                        pb.out[(long)m * NHID + n] = s;
                        pb.outb[(long)m * KPAD + n] = (__bf16)s;
                    }
                }
            }
}

// ---------------------------------------------------------------------------
// h_t = mean(states[1..8]); writes f32 h_t, bf16 h_t (for next stage A),
// and the final-hidden tail output on the last step
// ---------------------------------------------------------------------------
struct MeanArgs { const float* s[8]; float* out; __bf16* outb; float* out2; };

__global__ void mean_kernel(MeanArgs a) {
    const int i = blockIdx.x * 256 + threadIdx.x;
    if (i < BATCH * NHID) {
        float acc = 0.0f;
#pragma unroll
        for (int j = 0; j < 8; ++j) acc += a.s[j][i];
        acc *= 0.125f;
        a.out[i] = acc;
        const int r = i / NHID, c = i % NHID;
        a.outb[(long)r * KPAD + c] = (__bf16)acc;
        if (a.out2) a.out2[i] = acc;
    }
}

// ---------------------------------------------------------------------------
extern "C" void kernel_launch(void* const* d_in, const int* in_sizes, int n_in,
                              void* d_out, int out_size, void* d_ws, size_t ws_size,
                              hipStream_t stream) {
    (void)in_sizes; (void)n_in; (void)out_size; (void)ws_size;
    const float* X  = (const float*)d_in[0];   // [T,B,850]
    const float* H0 = (const float*)d_in[1];   // [1,B,850]
    const float* W0 = (const float*)d_in[2];   // [1700,1700]
    const float* Ws = (const float*)d_in[3];   // [8,850,1700]
    float* out = (float*)d_out;                // [T,B,850] ++ [1,B,850]

    // ---- workspace layout (all 16B-aligned) ----
    char*   ws   = (char*)d_ws;
    __bf16* wT   = (__bf16*)ws;                                  // 10*1700*864
    size_t  off  = 10ul * NTOT * KPAD * sizeof(__bf16);          // 29.4 MB
    __bf16* Xbf  = (__bf16*)(ws + off);                          // 32768*864
    off += (size_t)T_STEPS * BATCH * KPAD * sizeof(__bf16);      // 56.6 MB
    float*  Xproj = (float*)(ws + off);                          // 32768*1700
    off += (size_t)T_STEPS * BATCH * NTOT * sizeof(float);       // 222.8 MB
    __bf16* H0bf = (__bf16*)(ws + off);
    off += (size_t)BATCH * KPAD * sizeof(__bf16);
    __bf16* sbfB = (__bf16*)(ws + off);                          // s0..s8 bf16 + hbf
    off += 10ul * BATCH * KPAD * sizeof(__bf16);
    float*  S    = (float*)(ws + off);                           // s0..s8 f32
    float*  s[9]; __bf16* sb[9];
    for (int i = 0; i < 9; ++i) {
        s[i]  = S + (size_t)i * BATCH * NHID;
        sb[i] = sbfB + (size_t)i * BATCH * KPAD;
    }
    __bf16* hbf = sbfB + 9ul * BATCH * KPAD;

    const __bf16* W0xT = wT;
    const __bf16* W0hT = wT + (size_t)NTOT * KPAD;
    auto WsT = [&](int i) { return wT + (size_t)(2 + i) * NTOT * KPAD; };
    const size_t HOFF = (size_t)NHID * KPAD;   // offset of h-half rows in a BT

    // 1) one-time per-launch conversions (parallel, off critical path)
    {
        long t1 = 10L * NTOT * KPAD;
        convert_weights_kernel<<<(int)((t1 + 255) / 256), 256, 0, stream>>>(W0, Ws, wT);
        long t2 = (long)T_STEPS * BATCH * KPAD;
        cvt_pad_kernel<<<(int)((t2 + 255) / 256), 256, 0, stream>>>(Xbf, X, (long)T_STEPS * BATCH);
        long t3 = (long)BATCH * KPAD;
        cvt_pad_kernel<<<(int)((t3 + 255) / 256), 256, 0, stream>>>(H0bf, H0, BATCH);
        long t4 = 10L * BATCH * KPAD;   // zero state bf16 buffers (pads stay 0)
        cvt_pad_kernel<<<(int)((t4 + 255) / 256), 256, 0, stream>>>(sbfB, nullptr, 10L * BATCH);
    }

    // 2) bulk input projection for all T
    {
        ProbSet px{};
        px.p[0] = { Xbf, W0xT, nullptr, nullptr, nullptr, Xproj, nullptr, 3 };
        gemm_wmma_kernel<0><<<dim3((T_STEPS * BATCH) / BM, (NTOT + BN - 1) / BN, 1),
                              256, 0, stream>>>(px, NTOT);
    }

    // 3) sequential recurrence
    const dim3 g(BATCH / BM, (NHID + BN - 1) / BN, 1);
    for (int t = 0; t < T_STEPS; ++t) {
        const __bf16* hpb = (t == 0) ? H0bf : hbf;
        const float*  hpf = (t == 0) ? H0 : out + (size_t)(t - 1) * BATCH * NHID;

        // stage A: s0 = hp + sig(c0)*(tanh(h0)-hp)
        ProbSet pa{};
        pa.p[0] = { hpb, W0hT, W0hT + HOFF, hpf,
                    Xproj + (size_t)t * BATCH * NTOT, s[0], sb[0], 0 };
        gemm_wmma_kernel<1><<<dim3(g.x, g.y, 1), 256, 0, stream>>>(pa, NHID);

        // stage B: nodes 0,1 (pred s0; tanh, relu)
        ProbSet pbx{};
        pbx.p[0] = { sb[0], WsT(0), WsT(0) + HOFF, s[0], nullptr, s[1], sb[1], 0 };
        pbx.p[1] = { sb[0], WsT(1), WsT(1) + HOFF, s[0], nullptr, s[2], sb[2], 1 };
        gemm_wmma_kernel<1><<<dim3(g.x, g.y, 2), 256, 0, stream>>>(pbx, NHID);

        // stage C: nodes 2,3 (pred s1; sig,id), nodes 4,5 (pred s2; tanh,relu)
        ProbSet pc{};
        pc.p[0] = { sb[1], WsT(2), WsT(2) + HOFF, s[1], nullptr, s[3], sb[3], 2 };
        pc.p[1] = { sb[1], WsT(3), WsT(3) + HOFF, s[1], nullptr, s[4], sb[4], 3 };
        pc.p[2] = { sb[2], WsT(4), WsT(4) + HOFF, s[2], nullptr, s[5], sb[5], 0 };
        pc.p[3] = { sb[2], WsT(5), WsT(5) + HOFF, s[2], nullptr, s[6], sb[6], 1 };
        gemm_wmma_kernel<1><<<dim3(g.x, g.y, 4), 256, 0, stream>>>(pc, NHID);

        // stage D: node 6 (pred s3; sig), node 7 (pred s4; id)
        ProbSet pd{};
        pd.p[0] = { sb[3], WsT(6), WsT(6) + HOFF, s[3], nullptr, s[7], sb[7], 2 };
        pd.p[1] = { sb[4], WsT(7), WsT(7) + HOFF, s[4], nullptr, s[8], sb[8], 3 };
        gemm_wmma_kernel<1><<<dim3(g.x, g.y, 2), 256, 0, stream>>>(pd, NHID);

        // mean of states[1..8] -> h_t (f32 to d_out, bf16 for next step)
        MeanArgs ma{};
        for (int j = 0; j < 8; ++j) ma.s[j] = s[1 + j];
        ma.out  = out + (size_t)t * BATCH * NHID;
        ma.outb = hbf;
        ma.out2 = (t == T_STEPS - 1) ? out + (size_t)T_STEPS * BATCH * NHID : nullptr;
        mean_kernel<<<(BATCH * NHID + 255) / 256, 256, 0, stream>>>(ma);
    }
}
